// RobustPrompt_T_16372415333022
// MI455X (gfx1250) — matrix operand order repaired and checked by reading, not compiled, not run on hardware
//
#include <hip/hip_runtime.h>
#include <hip/hip_bf16.h>

typedef _Float16 v16h  __attribute__((ext_vector_type(16)));
typedef _Float16 half8 __attribute__((ext_vector_type(8)));
typedef _Float16 half4 __attribute__((ext_vector_type(4)));
typedef float    v8f   __attribute__((ext_vector_type(8)));

#define C_DIM 128
#define SIM_THR 0.2f
#define DEG_THR 5.0f

// ---------------------------------------------------------------------------
// Kernel 1: per-node L2 normalize, store as fp16 (halves gather traffic; the
// whole 25.6 MB table lives in the 192 MB L2 for the edge phase).
// One wave (32 lanes) per node, 4 floats per lane.
// ---------------------------------------------------------------------------
__global__ void normalize_fp16_kernel(const float* __restrict__ x,
                                      _Float16* __restrict__ xh, int N) {
    const int wave = (blockIdx.x * blockDim.x + threadIdx.x) >> 5;
    const int lane = threadIdx.x & 31;
    if (wave >= N) return;

    const float4 v = *(const float4*)(x + (size_t)wave * C_DIM + lane * 4);
    float s = v.x * v.x + v.y * v.y + v.z * v.z + v.w * v.w;
    #pragma unroll
    for (int m = 16; m >= 1; m >>= 1) s += __shfl_xor(s, m, 32);
    const float inv = 1.0f / sqrtf(s);

    half4 h;
    h[0] = (_Float16)(v.x * inv);
    h[1] = (_Float16)(v.y * inv);
    h[2] = (_Float16)(v.z * inv);
    h[3] = (_Float16)(v.w * inv);
    *(half4*)(xh + (size_t)wave * C_DIM + lane * 4) = h;
}

// ---------------------------------------------------------------------------
// Kernel 2: "all channels nonzero" flag for each prompt vector (reference's
// nz = all(record != 0)). One block of C_DIM threads, LDS atomic AND.
// ---------------------------------------------------------------------------
__global__ void prompt_flags_kernel(const float* __restrict__ p_sim,
                                    const float* __restrict__ p_deg,
                                    const float* __restrict__ p_other,
                                    int* __restrict__ flags) {
    __shared__ int s[3];
    const int t = threadIdx.x;
    if (t < 3) s[t] = 1;
    __syncthreads();
    if (p_sim[t]   == 0.0f) atomicAnd(&s[0], 0);
    if (p_deg[t]   == 0.0f) atomicAnd(&s[1], 0);
    if (p_other[t] == 0.0f) atomicAnd(&s[2], 0);
    __syncthreads();
    if (t < 3) flags[t] = s[t];
}

// ---------------------------------------------------------------------------
// Kernel 3: edge cosine sims via WMMA. One wave handles 16 edges:
//   A[i,k] = xh[row_i][k], B[k,j] = xh[col_j][k]   (K chunked 32-wide, 4x)
//   e_i = diag(A x B) accumulated in f32 by v_wmma_f32_16x16x32_f16.
// A layout (ISA 16-bit A 16x32): lane<16 -> K 0..7 & 16..23; lane>=16 -> K 8..15 & 24..31.
// B layout (ISA 16-bit B):       lane<16 -> K 0..7 & 16..23; lane>=16 -> K 8..15 & 24..31.
// C/D diag: (i,i) sits in acc[i] of lane i (i<8) and acc[l-24] of lane l=i+16 (i>=8).
// ---------------------------------------------------------------------------
__global__ void edge_wmma_kernel(const _Float16* __restrict__ xh,
                                 const int* __restrict__ edge_index,
                                 float* __restrict__ cbuf,
                                 float* __restrict__ degbuf,
                                 int E) {
    const int wave = (blockIdx.x * blockDim.x + threadIdx.x) >> 5;
    const int lane = threadIdx.x & 31;
    const int base = wave * 16;
    if (base >= E) return;                       // wave-uniform exit (EXEC stays full)

    // each half-wave loads the same 16 edges' endpoints (m = lane & 15)
    const int m   = lane & 15;
    const int eid = min(base + m, E - 1);        // clamp tail; padded lanes never scatter
    const int r   = edge_index[eid];             // row endpoints   [0..E)
    const int c   = edge_index[E + eid];         // col endpoints   [E..2E)

    const _Float16* pr = xh + (size_t)r * C_DIM;
    const _Float16* pc = xh + (size_t)c * C_DIM;
    const int kbA = (lane < 16) ? 0 : 8;         // A lane-half K base
    const int kbB = (lane < 16) ? 0 : 8;         // B lane-half K base

    v8f acc = {};
    #pragma unroll
    for (int kc = 0; kc < 4; ++kc) {
        const int kA = kc * 32 + kbA;
        const int kB = kc * 32 + kbB;
        half8 a0 = *(const half8*)(pr + kA);
        half8 a1 = *(const half8*)(pr + kA + 16);
        half8 b0 = *(const half8*)(pc + kB);
        half8 b1 = *(const half8*)(pc + kB + 16);
        v16h A = __builtin_shufflevector(a0, a1, 0,1,2,3,4,5,6,7,8,9,10,11,12,13,14,15);
        v16h B = __builtin_shufflevector(b0, b1, 0,1,2,3,4,5,6,7,8,9,10,11,12,13,14,15);
        acc = __builtin_amdgcn_wmma_f32_16x16x32_f16(
                  /*neg_a=*/false, A, /*neg_b=*/false, B,
                  /*c_mod=*/(short)0, acc, /*reuse_a=*/false, /*reuse_b=*/false);
    }

    // diagonal extraction + float-atomic scatter (16 active lanes per wave)
    const bool act = (lane < 8) || (lane >= 24);
    const int  g   = (lane < 8) ? lane : (lane - 24);
    float e = acc[0];
    if (g == 1) e = acc[1];
    if (g == 2) e = acc[2];
    if (g == 3) e = acc[3];
    if (g == 4) e = acc[4];
    if (g == 5) e = acc[5];
    if (g == 6) e = acc[6];
    if (g == 7) e = acc[7];
    if (act && (base + m) < E) {
        atomicAdd(cbuf + c, e);
        atomicAdd(degbuf + c, 1.0f);
    }
}

// ---------------------------------------------------------------------------
// Kernel 4: per-element mask/blend + residual add.
// ---------------------------------------------------------------------------
__global__ void finalize_kernel(const float* __restrict__ x,
                                const float* __restrict__ cbuf,
                                const float* __restrict__ degbuf,
                                const float* __restrict__ p_sim,
                                const float* __restrict__ p_deg,
                                const float* __restrict__ p_other,
                                const int* __restrict__ flags,
                                float* __restrict__ out, long total) {
    const long idx = (long)blockIdx.x * blockDim.x + threadIdx.x;
    if (idx >= total) return;
    const int n  = (int)(idx >> 7);      // C_DIM == 128
    const int ch = (int)(idx & 127);

    const float cn = cbuf[n];
    const float dn = degbuf[n];
    const float deg_safe = fmaxf(dn, 1.0f);
    const bool msim = (dn > 0.0f) && ((cn / deg_safe) <= SIM_THR);
    const bool mdeg = (dn <= DEG_THR);
    const bool moth = !(msim || mdeg);

    const int nlen = (msim && flags[0]) + (mdeg && flags[1]) + (moth && flags[2]);
    float sum = 0.0f;
    if (msim) sum += p_sim[ch];
    if (mdeg) sum += p_deg[ch];
    if (moth) sum += p_other[ch];
    const float fin = (nlen > 0) ? (sum / (float)nlen) : 0.0f;
    out[idx] = x[idx] + fin;
}

// ---------------------------------------------------------------------------
extern "C" void kernel_launch(void* const* d_in, const int* in_sizes, int n_in,
                              void* d_out, int out_size, void* d_ws, size_t ws_size,
                              hipStream_t stream) {
    const float* x          = (const float*)d_in[0];
    const int*   edge_index = (const int*)d_in[1];
    const float* p_sim      = (const float*)d_in[2];
    const float* p_deg      = (const float*)d_in[3];
    const float* p_other    = (const float*)d_in[4];
    float* out = (float*)d_out;

    const int Cn = in_sizes[2];              // 128
    const int N  = in_sizes[0] / Cn;         // 100000
    const int E  = in_sizes[1] / 2;          // 1600000

    // workspace layout
    char* wsb = (char*)d_ws;
    _Float16* xh = (_Float16*)wsb;                                   // N*C fp16
    size_t off = ((size_t)N * Cn * sizeof(_Float16) + 255) & ~(size_t)255;
    float* cbuf   = (float*)(wsb + off);                             // N f32
    float* degbuf = cbuf + N;                                        // N f32
    int*   flags  = (int*)(degbuf + N);                              // 3 int

    // zero the scatter accumulators (harness does not re-zero between replays)
    hipMemsetAsync(cbuf, 0, (size_t)2 * N * sizeof(float), stream);

    // 1) normalize -> fp16 table (wave per node)
    {
        const int waves_per_block = 8;                // 256 threads
        const int grid = (N + waves_per_block - 1) / waves_per_block;
        normalize_fp16_kernel<<<grid, waves_per_block * 32, 0, stream>>>(x, xh, N);
    }

    // 2) prompt nonzero flags
    prompt_flags_kernel<<<1, Cn, 0, stream>>>(p_sim, p_deg, p_other, flags);

    // 3) edge cosine sims via WMMA + atomic scatter (wave per 16 edges)
    {
        const int tiles = (E + 15) / 16;              // waves needed
        const int waves_per_block = 8;                // 256 threads
        const int grid = (tiles + waves_per_block - 1) / waves_per_block;
        edge_wmma_kernel<<<grid, waves_per_block * 32, 0, stream>>>(
            xh, edge_index, cbuf, degbuf, E);
    }

    // 4) finalize: masks + blend + residual
    {
        const long total = (long)N * Cn;
        const int block = 256;
        const long grid = (total + block - 1) / block;
        finalize_kernel<<<(int)grid, block, 0, stream>>>(
            x, cbuf, degbuf, p_sim, p_deg, p_other, flags, out, total);
    }
}
